// VisionMoEAdapter_67302137528986
// MI455X (gfx1250) — matrix operand order repaired and checked by reading, not compiled, not run on hardware
//
#include <hip/hip_runtime.h>
#include <hip/hip_bf16.h>

#define B_    4
#define S_    2048
#define D_    4096
#define VLEN_ 614
#define TPB   256
#define BM    128
#define BN    64
#define BK    32
#define ROWDW 20   // dwords per LDS row: 16 data dwords (32 halves) + 4 pad -> 80B, 16B aligned
#define ROWH  (ROWDW * 2)

typedef __attribute__((ext_vector_type(16))) _Float16 v16h;
typedef __attribute__((ext_vector_type(8)))  float    v8f;

union Frag { v16h v; unsigned int d[8]; };
union PackH { _Float16 h[2]; unsigned int u; };

// low 32 bits of a generic pointer to an LDS object == LDS byte offset
__device__ __forceinline__ unsigned lds_addr32(const void* p) {
  return (unsigned)(unsigned long long)p;
}
__device__ __forceinline__ void wait_asynccnt0() {
#if __has_builtin(__builtin_amdgcn_s_wait_asynccnt)
  __builtin_amdgcn_s_wait_asynccnt(0);
#else
  asm volatile("s_wait_asynccnt 0x0" ::: "memory");
#endif
}

// ---- stage 1: per-row sum of |h| -------------------------------------------
__global__ __launch_bounds__(TPB) void rowsum_kernel(const float* __restrict__ h,
                                                     float* __restrict__ rowsum) {
  __shared__ float sm[TPB];
  const int row = blockIdx.x;                 // 0 .. B*S-1
  const float* hr = h + (size_t)row * D_;
  float acc = 0.f;
  for (int k = threadIdx.x; k < D_; k += TPB) acc += fabsf(hr[k]);
  sm[threadIdx.x] = acc;
  __syncthreads();
  for (int o = TPB / 2; o > 0; o >>= 1) {
    if (threadIdx.x < o) sm[threadIdx.x] += sm[threadIdx.x + o];
    __syncthreads();
  }
  if (threadIdx.x == 0) rowsum[row] = sm[0];
}

// ---- stage 2: per-batch modality ratio -------------------------------------
__global__ __launch_bounds__(TPB) void ratio_kernel(const float* __restrict__ rowsum,
                                                    float* __restrict__ ratio) {
  __shared__ float sv[TPB], st[TPB];
  const int b = blockIdx.x;
  float v = 0.f, t = 0.f;
  for (int s = threadIdx.x; s < S_; s += TPB) {
    float r = rowsum[b * S_ + s];
    if (s < VLEN_) v += r; else t += r;
  }
  sv[threadIdx.x] = v; st[threadIdx.x] = t;
  __syncthreads();
  for (int o = TPB / 2; o > 0; o >>= 1) {
    if (threadIdx.x < o) { sv[threadIdx.x] += sv[threadIdx.x + o];
                           st[threadIdx.x] += st[threadIdx.x + o]; }
    __syncthreads();
  }
  if (threadIdx.x == 0) {
    float vis = sv[0] / ((float)VLEN_ * (float)D_);
    float txt = st[0] / ((float)(S_ - VLEN_) * (float)D_);
    ratio[b] = txt / (vis + 1e-8f);
  }
}

// ---- gating: softmax-3, top-2 per token; indices from token 0 --------------
__global__ __launch_bounds__(TPB) void gate_kernel(const float* __restrict__ h,
                                                   const float* __restrict__ gw,
                                                   const float* __restrict__ gb,
                                                   const float* __restrict__ ratio,
                                                   float* __restrict__ wtok,
                                                   int* __restrict__ idxb) {
  __shared__ float sm[3 * TPB];
  const int token = blockIdx.x;
  const int b = token / S_, s = token % S_;
  const float* hrow = h + (size_t)token * D_;
  float l0 = 0.f, l1 = 0.f, l2 = 0.f;
  for (int k = threadIdx.x; k < D_; k += TPB) {
    float hv = hrow[k];
    l0 += hv * gw[k * 3 + 0];
    l1 += hv * gw[k * 3 + 1];
    l2 += hv * gw[k * 3 + 2];
  }
  sm[threadIdx.x] = l0; sm[TPB + threadIdx.x] = l1; sm[2 * TPB + threadIdx.x] = l2;
  __syncthreads();
  for (int o = TPB / 2; o > 0; o >>= 1) {
    if (threadIdx.x < o) {
      sm[threadIdx.x]           += sm[threadIdx.x + o];
      sm[TPB + threadIdx.x]     += sm[TPB + threadIdx.x + o];
      sm[2 * TPB + threadIdx.x] += sm[2 * TPB + threadIdx.x + o];
    }
    __syncthreads();
  }
  if (threadIdx.x == 0) {
    const float r = ratio[b];
    float lg[3];
    lg[0] = sm[0]       + r * gw[(size_t)D_ * 3 + 0] + gb[0];
    lg[1] = sm[TPB]     + r * gw[(size_t)D_ * 3 + 1] + gb[1];
    lg[2] = sm[2 * TPB] + r * gw[(size_t)D_ * 3 + 2] + gb[2];
    float m = fmaxf(lg[0], fmaxf(lg[1], lg[2]));
    float e0 = __expf(lg[0] - m), e1 = __expf(lg[1] - m), e2 = __expf(lg[2] - m);
    float inv = 1.0f / (e0 + e1 + e2);
    float sc[3] = { e0 * inv, e1 * inv, e2 * inv };
    int i0 = 0; float v0 = sc[0];
    if (sc[1] > v0) { v0 = sc[1]; i0 = 1; }
    if (sc[2] > v0) { v0 = sc[2]; i0 = 2; }
    int i1 = -1; float v1 = -1.0f;
    for (int e = 0; e < 3; e++) { if (e == i0) continue; if (sc[e] > v1) { v1 = sc[e]; i1 = e; } }
    float inv2 = 1.0f / (v0 + v1 + 1e-8f);
    wtok[(size_t)token * 2 + 0] = v0 * inv2;
    wtok[(size_t)token * 2 + 1] = v1 * inv2;
    if (s == 0) { idxb[b * 2 + 0] = i0; idxb[b * 2 + 1] = i1; }
  }
}

// ---- WMMA fragment loaders (contiguous k = 16*kh + s in both A and B); 16B-aligned runs ------
__device__ __forceinline__ void load_fragA(const unsigned int* lsA, int row, int kh, Frag& f) {
  const unsigned int* base = lsA + row * ROWDW;
  #pragma unroll
  for (int j = 0; j < 8; j++) f.d[j] = base[kh * 8 + j];            // K = kh*16 + 2j
}
__device__ __forceinline__ void load_fragB(const unsigned int* lsB, int col, int kh, Frag& f) {
  const unsigned int* base = lsB + col * ROWDW;                     // B stored [n][k]
  #pragma unroll
  for (int j = 0; j < 8; j++) f.d[j] = base[kh * 8 + j];            // K = kh*16 + 2j
}

// B-tile stage: thread handles 2(k) x 4(n); packs (k,k+1) halves -> ds_store_b32
__device__ __forceinline__ void stage_B(const float* __restrict__ wblk, int k0,
                                        unsigned int* lsB, int tid) {
  const int k2 = tid >> 4;            // 0..15 -> k pair
  const int n4 = (tid & 15) * 4;      // 0..60
  const float* s0 = wblk + (size_t)(k0 + 2 * k2) * D_ + n4;
  const float* s1 = s0 + D_;
  #pragma unroll
  for (int j = 0; j < 4; j++) {
    PackH p;
    p.h[0] = (_Float16)s0[j];
    p.h[1] = (_Float16)s1[j];
    lsB[(n4 + j) * ROWDW + k2] = p.u;
  }
  if (k0 + BK < D_) __builtin_prefetch(s0 + (size_t)BK * D_, 0, 3);
}

// ---- GEMM1: act[b,slot] = f_e(h[b] @ w1[e] + b1[e]),  e = idxb[b,slot] ------
__global__ __launch_bounds__(TPB) void gemm1_kernel(const float* __restrict__ h,
                                                    const float* __restrict__ w1,
                                                    const float* __restrict__ b1,
                                                    const int* __restrict__ idxb,
                                                    _Float16* __restrict__ act) {
  __shared__ unsigned int lsA[BM * ROWDW];
  __shared__ unsigned int lsB[BN * ROWDW];
  _Float16* lsAh = (_Float16*)lsA;

  const int tid = threadIdx.x;
  const int bm = blockIdx.x, bn = blockIdx.y;
  const int b = blockIdx.z >> 1, slot = blockIdx.z & 1;
  const int e = idxb[b * 2 + slot];

  const float* hblk = h  + ((size_t)b * S_ + (size_t)bm * BM) * D_;
  const float* w1e  = w1 + (size_t)e * D_ * D_ + (size_t)bn * BN;

  const int wid = tid >> 5, lane = tid & 31;
  const int waveM = wid >> 1, waveN = wid & 1;
  const int kh = lane >> 4, lmod = lane & 15;
  const int ar = tid >> 1, ac = (tid & 1) * 16;       // A loader: 2 thr/row, 16 floats each

  v8f acc[2][2] = {};

  for (int k0 = 0; k0 < D_; k0 += BK) {
    __syncthreads();
    {
      const float* sA = hblk + (size_t)ar * D_ + k0 + ac;
      _Float16* dA = lsAh + ar * ROWH + ac;
      #pragma unroll
      for (int i = 0; i < 16; i++) dA[i] = (_Float16)sA[i];
      if (k0 + BK < D_) __builtin_prefetch(sA + BK, 0, 3);
      stage_B(w1e, k0, lsB, tid);
    }
    __syncthreads();

    Frag fa[2], fb[2];
    #pragma unroll
    for (int mi = 0; mi < 2; mi++) load_fragA(lsA, waveM * 32 + mi * 16 + lmod, kh, fa[mi]);
    #pragma unroll
    for (int ni = 0; ni < 2; ni++) load_fragB(lsB, waveN * 32 + ni * 16 + lmod, kh, fb[ni]);
    #pragma unroll
    for (int mi = 0; mi < 2; mi++)
      #pragma unroll
      for (int ni = 0; ni < 2; ni++)
        acc[mi][ni] = __builtin_amdgcn_wmma_f32_16x16x32_f16(
            false, fa[mi].v, false, fb[ni].v, (short)0, acc[mi][ni], false, false);
  }

  _Float16* actOut = act + (size_t)(b * 2 + slot) * S_ * D_;
  #pragma unroll
  for (int mi = 0; mi < 2; mi++)
    #pragma unroll
    for (int ni = 0; ni < 2; ni++)
      #pragma unroll
      for (int r = 0; r < 8; r++) {
        const int sG = bm * BM + waveM * 32 + mi * 16 + kh * 8 + r;
        const int nG = bn * BN + waveN * 32 + ni * 16 + lmod;
        float v = acc[mi][ni][r] + b1[e * D_ + nG];
        if (e == 0)      v = tanhf(v);
        else if (e == 1) v = 0.5f * v * (1.0f + erff(v * 0.70710678118654752f));
        else             v = v / (1.0f + __expf(-v));
        actOut[(size_t)sG * D_ + nG] = (_Float16)v;
      }
}

// ---- GEMM2: out[b] = sum_slot wtok[:,slot] * (act[b,slot] @ w2[e] + b2[e]) --
// A operand is already f16 in global ws -> async copy straight into LDS.
__global__ __launch_bounds__(TPB) void gemm2_kernel(const _Float16* __restrict__ act,
                                                    const float* __restrict__ w2,
                                                    const float* __restrict__ b2,
                                                    const int* __restrict__ idxb,
                                                    const float* __restrict__ wtok,
                                                    float* __restrict__ out) {
  __shared__ unsigned int lsA[BM * ROWDW];
  __shared__ unsigned int lsB[BN * ROWDW];
  _Float16* lsAh = (_Float16*)lsA;

  const int tid = threadIdx.x;
  const int bm = blockIdx.x, bn = blockIdx.y;
  const int b = blockIdx.z;

  const int wid = tid >> 5, lane = tid & 31;
  const int waveM = wid >> 1, waveN = wid & 1;
  const int kh = lane >> 4, lmod = lane & 15;
  const int ar = tid >> 1, ac = (tid & 1) * 16;

  const unsigned ldsA_dst = lds_addr32(lsAh + ar * ROWH + ac);  // 16B aligned (80B rows)

  float outv[2][2][8] = {};

  for (int slot = 0; slot < 2; ++slot) {
    const int e = idxb[b * 2 + slot];
    const _Float16* ablk = act + ((size_t)(b * 2 + slot) * S_ + (size_t)bm * BM) * D_;
    const float* w2e = w2 + (size_t)e * D_ * D_ + (size_t)bn * BN;

    v8f acc[2][2] = {};
    for (int k0 = 0; k0 < D_; k0 += BK) {
      __syncthreads();
      {
        // async f16 tile copy: global -> LDS, tracked by ASYNCcnt
        const _Float16* gA = ablk + (size_t)ar * D_ + k0 + ac;     // 32B aligned
        asm volatile("global_load_async_to_lds_b128 %0, %1, off"
                     :: "v"(ldsA_dst), "v"(gA) : "memory");
        asm volatile("global_load_async_to_lds_b128 %0, %1, off offset:16"
                     :: "v"(ldsA_dst), "v"(gA) : "memory");
        if (k0 + BK < D_) __builtin_prefetch(gA + BK, 0, 3);
        stage_B(w2e, k0, lsB, tid);
      }
      wait_asynccnt0();
      __syncthreads();

      Frag fa[2], fb[2];
      #pragma unroll
      for (int mi = 0; mi < 2; mi++) load_fragA(lsA, waveM * 32 + mi * 16 + lmod, kh, fa[mi]);
      #pragma unroll
      for (int ni = 0; ni < 2; ni++) load_fragB(lsB, waveN * 32 + ni * 16 + lmod, kh, fb[ni]);
      #pragma unroll
      for (int mi = 0; mi < 2; mi++)
        #pragma unroll
        for (int ni = 0; ni < 2; ni++)
          acc[mi][ni] = __builtin_amdgcn_wmma_f32_16x16x32_f16(
              false, fa[mi].v, false, fb[ni].v, (short)0, acc[mi][ni], false, false);
    }

    #pragma unroll
    for (int mi = 0; mi < 2; mi++)
      #pragma unroll
      for (int ni = 0; ni < 2; ni++)
        #pragma unroll
        for (int r = 0; r < 8; r++) {
          const int sG = bm * BM + waveM * 32 + mi * 16 + kh * 8 + r;
          const int nG = bn * BN + waveN * 32 + ni * 16 + lmod;
          float coef = wtok[((size_t)b * S_ + sG) * 2 + slot];
          outv[mi][ni][r] += coef * (acc[mi][ni][r] + b2[e * D_ + nG]);
        }
  }

  #pragma unroll
  for (int mi = 0; mi < 2; mi++)
    #pragma unroll
    for (int ni = 0; ni < 2; ni++)
      #pragma unroll
      for (int r = 0; r < 8; r++) {
        const int sG = bm * BM + waveM * 32 + mi * 16 + kh * 8 + r;
        const int nG = bn * BN + waveN * 32 + ni * 16 + lmod;
        out[((size_t)b * S_ + sG) * D_ + nG] = outv[mi][ni][r];
      }
}

extern "C" void kernel_launch(void* const* d_in, const int* in_sizes, int n_in,
                              void* d_out, int out_size, void* d_ws, size_t ws_size,
                              hipStream_t stream) {
  (void)in_sizes; (void)n_in; (void)out_size; (void)ws_size;
  const float* h  = (const float*)d_in[0];
  const float* gw = (const float*)d_in[1];
  const float* gb = (const float*)d_in[2];
  const float* w1 = (const float*)d_in[3];
  const float* b1 = (const float*)d_in[4];
  const float* w2 = (const float*)d_in[5];
  const float* b2 = (const float*)d_in[6];
  float* out = (float*)d_out;

  char* ws = (char*)d_ws;
  size_t off = 0;
  auto take = [&](size_t bytes) -> char* {
    char* p = ws + off;
    off = (off + bytes + 255) & ~(size_t)255;
    return p;
  };
  _Float16* act   = (_Float16*)take((size_t)B_ * 2 * S_ * D_ * sizeof(_Float16)); // 128 MB
  float*    rsum  = (float*)take((size_t)B_ * S_ * sizeof(float));
  float*    ratio = (float*)take((size_t)B_ * sizeof(float));
  float*    wtok  = (float*)take((size_t)B_ * S_ * 2 * sizeof(float));
  int*      idxb  = (int*)take((size_t)B_ * 2 * sizeof(int));

  rowsum_kernel<<<B_ * S_, TPB, 0, stream>>>(h, rsum);
  ratio_kernel <<<B_,      TPB, 0, stream>>>(rsum, ratio);
  gate_kernel  <<<B_ * S_, TPB, 0, stream>>>(h, gw, gb, ratio, wtok, idxb);
  gemm1_kernel <<<dim3(S_ / BM, D_ / BN, B_ * 2), TPB, 0, stream>>>(h, w1, b1, idxb, act);
  gemm2_kernel <<<dim3(S_ / BM, D_ / BN, B_),     TPB, 0, stream>>>(act, w2, b2, idxb, wtok, out);
}
